// NeRF_15908558864502
// MI455X (gfx1250) — matrix operand order repaired and checked by reading, not compiled, not run on hardware
//
#include <hip/hip_runtime.h>

// ---------------------------------------------------------------------------
// NeRF fused MLP for gfx1250 (CDNA5, wave32, WMMA f32_16x16x32_f16)
// M = 64 rows per block to double weight reuse out of L2.
// ---------------------------------------------------------------------------

typedef __attribute__((ext_vector_type(16))) _Float16 v16h;
typedef __attribute__((ext_vector_type(8)))  _Float16 v8h;
typedef __attribute__((ext_vector_type(8)))  float    v8f;

#define LSTRIDE 328   // halves; 656B row stride -> 164 DWORDs, spreads LDS banks
#define MROWS   64    // rows per block

// Packed-weight offsets in f16 elements inside d_ws.
// Layer K is padded to a multiple of 32; N is a multiple of 16.
// Fragment layout: frag = nt*KB + kb ; within frag: lane*16 + (2*i + h)
//   k = kb*32 + (lane>=16 ? 16 : 0) + 2*i + h ; n = nt*16 + (lane&15)
static constexpr size_t OFF0  = 0;                      //  64x256
static constexpr size_t OFF1  = OFF0  + 64  * 256;      // 256x256
static constexpr size_t OFF2  = OFF1  + 256 * 256;
static constexpr size_t OFF3  = OFF2  + 256 * 256;
static constexpr size_t OFF4  = OFF3  + 256 * 256;      // 320x256 (skip layer)
static constexpr size_t OFF5  = OFF4  + 320 * 256;
static constexpr size_t OFF6  = OFF5  + 256 * 256;
static constexpr size_t OFF7  = OFF6  + 256 * 256;
static constexpr size_t OFFF  = OFF7  + 256 * 256;      // final 256x256
static constexpr size_t OFFR1 = OFFF  + 256 * 256;      // rgb1 288x128
static constexpr size_t WP_TOTAL = OFFR1 + 288 * 128;   // 593920 halves = 1.19 MB

// ---------------------------------------------------------------------------
// Weight pre-pack: fp32 [K,N] row-major -> f16 wave32 B-fragment layout.
// skipGap=1 applies the layer-4 remap: LDS col 63 is a zero pad, cols >=64
// map to weight row (k-1)  (concat([xyz(63), h(256)]) input).
// ---------------------------------------------------------------------------
__global__ __launch_bounds__(256) void pack_weights(
    const float* __restrict__ src, _Float16* __restrict__ dst,
    int KB, int N, int Kact, int skipGap, int total)
{
  const int e = blockIdx.x * 256 + threadIdx.x;
  if (e >= total) return;
  const int frag = e >> 9;          // 512 halves per 32x16 fragment
  const int rem  = e & 511;
  const int l    = rem >> 4;        // lane 0..31
  const int q    = rem & 15;        // 2*i + h
  const int kb   = frag % KB;
  const int nt   = frag / KB;
  const int k    = kb * 32 + ((l >> 4) << 4) + q;
  const int n    = nt * 16 + (l & 15);
  float v = 0.0f;
  if (skipGap) {
    if (k < 63)       v = src[k * N + n];
    else if (k >= 64) v = src[(k - 1) * N + n];
  } else if (k < Kact) {
    v = src[(size_t)k * N + n];
  }
  dst[e] = (_Float16)v;
}

// ---------------------------------------------------------------------------
// Wave-level tiled GEMM pass: one 16-row M-tile x NTPW consecutive N-tiles.
// A fragment (f16 16x32 layout) read from LDS, B fragments from packed global
// weights (one contiguous 32B read per lane per tile), bias+ReLU fused.
// ---------------------------------------------------------------------------
template <int KB, int NTPW, bool RELU>
__device__ __forceinline__ void gemm_pass(
    const _Float16* __restrict__ in, int inBase,
    _Float16* __restrict__ outLds, int outBase,
    const _Float16* __restrict__ wp,
    const float* __restrict__ bias,
    int mt, int ntBase, int lane)
{
  const int hi   = lane >> 4;     // 0: lanes 0-15, 1: lanes 16-31
  const int lm   = lane & 15;
  const int rowA = mt * 16 + lm;

  v8f acc[NTPW];
#pragma unroll
  for (int j = 0; j < NTPW; ++j) acc[j] = {};

#pragma unroll
  for (int kb = 0; kb < KB; ++kb) {
    // A fragment: lanes 0-15 hold K{0..15}, lanes 16-31 K{16..31}.
    const int c0 = inBase + kb * 32 + hi * 16;
    const v8h a0 = *(const v8h*)&in[rowA * LSTRIDE + c0];
    const v8h a1 = *(const v8h*)&in[rowA * LSTRIDE + c0 + 8];
    const v16h a = __builtin_shufflevector(a0, a1,
        0, 1, 2, 3, 4, 5, 6, 7, 8, 9, 10, 11, 12, 13, 14, 15);
#pragma unroll
    for (int j = 0; j < NTPW; ++j) {
      const int nt = ntBase + j;
      const v16h b = *(const v16h*)(wp + (((size_t)(nt * KB + kb)) << 9) + (lane << 4));
      acc[j] = __builtin_amdgcn_wmma_f32_16x16x32_f16(
          false, a, false, b, (short)0, acc[j], false, false);
    }
  }

  // D layout: VGPR r -> (M = r + hi*8, N = lm) within the 16x16 tile.
#pragma unroll
  for (int j = 0; j < NTPW; ++j) {
    const int nt = ntBase + j;
    const int n  = nt * 16 + lm;
    const float bv = bias[n];
#pragma unroll
    for (int r = 0; r < 8; ++r) {
      float v = acc[j][r] + bv;
      if (RELU) v = fmaxf(v, 0.0f);
      outLds[(mt * 16 + hi * 8 + r) * LSTRIDE + outBase + n] = (_Float16)v;
    }
  }
}

// Full layer: 4 M-tiles x 16 N-tiles, 8 waves. wave -> mt = wave&3,
// N-tile half = wave>>2; each wave runs 2 passes of 4 N-tiles (NT=16)
// or 1 pass of 4 (NT=8, rgb1).
template <int KB, int NT, bool RELU>
__device__ __forceinline__ void gemm_layer(
    const _Float16* __restrict__ in, int inBase,
    _Float16* __restrict__ outLds, int outBase,
    const _Float16* __restrict__ wp,
    const float* __restrict__ bias,
    int wave, int lane)
{
  const int mt = wave & 3;
  const int hf = wave >> 2;          // 0 or 1
#pragma unroll
  for (int g = 0; g < NT / 8; ++g)
    gemm_pass<KB, 4, RELU>(in, inBase, outLds, outBase, wp, bias,
                           mt, hf * (NT / 2) + g * 4, lane);
}

struct NerfParams {
  const float* x;
  const _Float16* wp;
  const float* bias[10];   // b0..b7, b_final, b_rgb1
  const float* w_sigma;
  const float* b_sigma;
  const float* w_rgb2;
  const float* b_rgb2;
  float* out;
};

// ---------------------------------------------------------------------------
// Fully fused NeRF forward: 64 rows per block, 8 wave32s, activations in LDS.
// Buffer layout: cols 0..63 hold posenc(xyz) (persisted for the skip concat),
// layer outputs live in cols 64..319 and ping-pong between bufA/bufB.
// ---------------------------------------------------------------------------
__global__ __launch_bounds__(256) void nerf_fused(NerfParams p)
{
  __shared__ alignas(32) _Float16 bufA[MROWS * LSTRIDE];
  __shared__ alignas(32) _Float16 bufB[MROWS * LSTRIDE];
  __shared__ alignas(16) _Float16 dirE[MROWS * 32];
  __shared__ float sigmaS[MROWS];

  const int base = blockIdx.x * MROWS;
  const int tid  = threadIdx.x;
  const int wave = tid >> 5;     // wave32
  const int lane = tid & 31;

  // ---- positional encodings (nerf-pl order: per freq [s0 s1 s2 c0 c1 c2]) --
  for (int idx = tid; idx < MROWS * 64; idx += 256) {
    const int row = idx >> 6, col = idx & 63;
    const float* xr = p.x + (size_t)(base + row) * 6;
    float v = 0.0f;
    if (col < 3) v = xr[col];
    else if (col < 63) {
      const int c = col - 3, f = c / 6, rr = c % 6;
      const float arg = xr[rr % 3] * (float)(1 << f);
      v = (rr < 3) ? __sinf(arg) : __cosf(arg);
    }
    bufA[row * LSTRIDE + col] = (_Float16)v;     // col 63 zero pad
  }
  for (int idx = tid; idx < MROWS * 32; idx += 256) {
    const int row = idx >> 5, col = idx & 31;
    const float* xr = p.x + (size_t)(base + row) * 6 + 3;
    float v = 0.0f;
    if (col < 3) v = xr[col];
    else if (col < 27) {
      const int c = col - 3, f = c / 6, rr = c % 6;
      const float arg = xr[rr % 3] * (float)(1 << f);
      v = (rr < 3) ? __sinf(arg) : __cosf(arg);
    }
    dirE[row * 32 + col] = (_Float16)v;          // cols 27..31 zero pad
  }
  // warm next layer's weights while posenc finishes
  __builtin_prefetch(p.wp + OFF0 + (tid << 6), 0, 0);
  __syncthreads();

  // ---- 8 FC layers (skip concat at layer 4 via col layout + remapped w4) ---
  gemm_layer<2,  16, true >(bufA, 0,  bufB, 64, p.wp + OFF1 - OFF1 + OFF0, p.bias[0], wave, lane);
  __builtin_prefetch(p.wp + OFF1 + (tid << 6), 0, 0);
  __syncthreads();
  gemm_layer<8,  16, true >(bufB, 64, bufA, 64, p.wp + OFF1, p.bias[1], wave, lane);
  __builtin_prefetch(p.wp + OFF2 + (tid << 6), 0, 0);
  __syncthreads();
  gemm_layer<8,  16, true >(bufA, 64, bufB, 64, p.wp + OFF2, p.bias[2], wave, lane);
  __builtin_prefetch(p.wp + OFF3 + (tid << 6), 0, 0);
  __syncthreads();
  gemm_layer<8,  16, true >(bufB, 64, bufA, 64, p.wp + OFF3, p.bias[3], wave, lane);
  __builtin_prefetch(p.wp + OFF4 + (tid << 6), 0, 0);
  __syncthreads();
  gemm_layer<10, 16, true >(bufA, 0,  bufB, 64, p.wp + OFF4, p.bias[4], wave, lane);
  __builtin_prefetch(p.wp + OFF5 + (tid << 6), 0, 0);
  __syncthreads();
  gemm_layer<8,  16, true >(bufB, 64, bufA, 64, p.wp + OFF5, p.bias[5], wave, lane);
  __builtin_prefetch(p.wp + OFF6 + (tid << 6), 0, 0);
  __syncthreads();
  gemm_layer<8,  16, true >(bufA, 64, bufB, 64, p.wp + OFF6, p.bias[6], wave, lane);
  __builtin_prefetch(p.wp + OFF7 + (tid << 6), 0, 0);
  __syncthreads();
  gemm_layer<8,  16, true >(bufB, 64, bufA, 64, p.wp + OFF7, p.bias[7], wave, lane);
  __builtin_prefetch(p.wp + OFFF + (tid << 6), 0, 0);
  __syncthreads();
  // h (post-ReLU) now in bufA cols 64..319

  // final (no ReLU): bufA[64..319] -> bufB[0..255]
  gemm_layer<8,  16, false>(bufA, 64, bufB, 0, p.wp + OFFF, p.bias[8], wave, lane);
  __builtin_prefetch(p.wp + OFFR1 + (tid << 6), 0, 0);

  // sigma head on waves 0-1 (reads bufA, which `final` does not modify)
  if (tid < MROWS) {
    const int row = tid;
    float acc = p.b_sigma[0];
    for (int c = 0; c < 256; ++c)
      acc += (float)bufA[row * LSTRIDE + 64 + c] * p.w_sigma[c];
    sigmaS[row] = acc;
  }
  // concat dir encoding into bufB cols 256..287 (283..287 stay zero)
  for (int idx = tid; idx < MROWS * 32; idx += 256) {
    const int row = idx >> 5, col = idx & 31;
    bufB[row * LSTRIDE + 256 + col] = dirE[row * 32 + col];
  }
  __syncthreads();

  // rgb1: [64 x 288] x [288 x 128] + ReLU -> bufA cols 0..127
  gemm_layer<9,  8, true >(bufB, 0, bufA, 0, p.wp + OFFR1, p.bias[9], wave, lane);
  __syncthreads();

  // rgb2 (128 -> 3) + sigmoid, emit [rgb, sigma]
  if (tid < MROWS) {
    const int row = tid;
    float rgb[3];
#pragma unroll
    for (int k = 0; k < 3; ++k) rgb[k] = p.b_rgb2[k];
    for (int c = 0; c < 128; ++c) {
      const float h = (float)bufA[row * LSTRIDE + c];
#pragma unroll
      for (int k = 0; k < 3; ++k) rgb[k] += h * p.w_rgb2[c * 3 + k];
    }
    float* o = p.out + (size_t)(base + row) * 4;
#pragma unroll
    for (int k = 0; k < 3; ++k) o[k] = 1.0f / (1.0f + __expf(-rgb[k]));
    o[3] = sigmaS[row];
  }
}

// ---------------------------------------------------------------------------
extern "C" void kernel_launch(void* const* d_in, const int* in_sizes, int n_in,
                              void* d_out, int out_size, void* d_ws, size_t ws_size,
                              hipStream_t stream)
{
  (void)in_sizes; (void)n_in; (void)out_size; (void)ws_size;

  const float* x = (const float*)d_in[0];
  const float* w[8]; const float* b[8];
  for (int i = 0; i < 8; ++i) {
    w[i] = (const float*)d_in[1 + 2 * i];
    b[i] = (const float*)d_in[2 + 2 * i];
  }
  const float* w_final = (const float*)d_in[17];
  const float* b_final = (const float*)d_in[18];
  const float* w_sigma = (const float*)d_in[19];
  const float* b_sigma = (const float*)d_in[20];
  const float* w_rgb1  = (const float*)d_in[21];
  const float* b_rgb1  = (const float*)d_in[22];
  const float* w_rgb2  = (const float*)d_in[23];
  const float* b_rgb2  = (const float*)d_in[24];

  _Float16* wp = (_Float16*)d_ws;   // 1.19 MB of packed f16 weights

  auto pack = [&](const float* src, size_t off, int KB, int N, int Kact, int gap) {
    const int total = KB * 32 * N;
    pack_weights<<<(total + 255) / 256, 256, 0, stream>>>(
        src, wp + off, KB, N, Kact, gap, total);
  };
  pack(w[0],    OFF0,  2,  256, 63,  0);
  pack(w[1],    OFF1,  8,  256, 256, 0);
  pack(w[2],    OFF2,  8,  256, 256, 0);
  pack(w[3],    OFF3,  8,  256, 256, 0);
  pack(w[4],    OFF4,  10, 256, 319, 1);   // skip-gap remap
  pack(w[5],    OFF5,  8,  256, 256, 0);
  pack(w[6],    OFF6,  8,  256, 256, 0);
  pack(w[7],    OFF7,  8,  256, 256, 0);
  pack(w_final, OFFF,  8,  256, 256, 0);
  pack(w_rgb1,  OFFR1, 9,  128, 283, 0);

  NerfParams p;
  p.x = x; p.wp = wp;
  for (int i = 0; i < 8; ++i) p.bias[i] = b[i];
  p.bias[8] = b_final; p.bias[9] = b_rgb1;
  p.w_sigma = w_sigma; p.b_sigma = b_sigma;
  p.w_rgb2  = w_rgb2;  p.b_rgb2  = b_rgb2;
  p.out = (float*)d_out;

  nerf_fused<<<262144 / MROWS, 256, 0, stream>>>(p);
}
